// STORN_22101901705967
// MI455X (gfx1250) — compile-verified, matrix operand order checked by reading, not run on hardware
//
#include <hip/hip_runtime.h>
#include <math.h>

// ---------------------------------------------------------------------------
// STORN forward for MI455X (gfx1250, wave32, WMMA bf16 -> f32 accumulate)
// L=512, B=1024, K=64, IN=OBS=ST=32
// ---------------------------------------------------------------------------

typedef __bf16 bf16;
typedef __attribute__((ext_vector_type(16))) __bf16 v16bf;
typedef __attribute__((ext_vector_type(8)))  float  v8f;
typedef __attribute__((ext_vector_type(4)))  int    v4i;

#if defined(__has_builtin)
#  if __has_builtin(__builtin_amdgcn_global_load_async_to_lds_b128) && \
      __has_builtin(__builtin_amdgcn_s_wait_asynccnt)
#    define HAVE_ASYNC_LDS 1
#  endif
#endif
#ifndef HAVE_ASYNC_LDS
#  define HAVE_ASYNC_LDS 0
#endif

#if HAVE_ASYNC_LDS
typedef __attribute__((address_space(1))) v4i* gv4i_p;  // global int4*
typedef __attribute__((address_space(3))) v4i* lv4i_p;  // LDS int4*
#endif

static __device__ __forceinline__ v8f wmma_bf16(v16bf a, v16bf b, v8f c) {
  // D = A(16x32) * B(32x16) + C(16x16), f32 accum
  return __builtin_amdgcn_wmma_f32_16x16x32_bf16(
      /*neg_a=*/false, a, /*neg_b=*/false, b,
      /*c_mod=*/(short)0, c, /*reuse_a=*/false, /*reuse_b=*/false);
}

// 16-bit A-frag layout (ISA 7.12.2): lane m = lane&15; khalf = (lane>>4)*8.
// Elements 0..7  hold K = k0+khalf .. k0+khalf+7
// Elements 8..15 hold K = k0+khalf+16 .. k0+khalf+23
// B-frag (B = W^T, W row-major [N,K]): identical addressing with row = N.
static __device__ __forceinline__ v16bf ld_frag_f32(const float* __restrict__ src,
                                                    int ld, int row0, int k0) {
  const int lane = threadIdx.x & 31;
  const float* q = src + (size_t)(row0 + (lane & 15)) * ld + (k0 + ((lane >> 4) << 3));
  v16bf f;
#pragma unroll
  for (int i = 0; i < 8; ++i) f[i] = (bf16)q[i];
#pragma unroll
  for (int i = 0; i < 8; ++i) f[8 + i] = (bf16)q[16 + i];
  return f;
}

static __device__ __forceinline__ v16bf ld_frag_ldsf(const float* src, int ld,
                                                     int row0, int k0) {
  const int lane = threadIdx.x & 31;
  const float* q = src + (row0 + (lane & 15)) * ld + (k0 + ((lane >> 4) << 3));
  v16bf f;
#pragma unroll
  for (int i = 0; i < 8; ++i) f[i] = (bf16)q[i];
#pragma unroll
  for (int i = 0; i < 8; ++i) f[8 + i] = (bf16)q[16 + i];
  return f;
}

static __device__ __forceinline__ v16bf ld_frag_ldsbf(const bf16* src, int ld,
                                                      int row0, int k0) {
  const int lane = threadIdx.x & 31;
  const bf16* q = src + (row0 + (lane & 15)) * ld + (k0 + ((lane >> 4) << 3));
  v16bf f;
#pragma unroll
  for (int i = 0; i < 8; ++i) f[i] = q[i];
#pragma unroll
  for (int i = 0; i < 8; ++i) f[8 + i] = q[16 + i];
  return f;
}

static __device__ __forceinline__ float sigmoidf_(float x) {
  return 1.0f / (1.0f + __expf(-x));
}

// ---------------------------------------------------------------------------
// Generic tiled GEMM: out[M,N] = act(in[M,K] @ W^T + bias) (optional accumulate)
// W row-major [N, ldw], K read from W cols [0,K). One wave per 16-row tile.
// Weight panel W[N,K] is staged once per block into LDS (async global->LDS on
// gfx1250 when available; ASYNCcnt-tracked), then all 8 waves read B-frags
// from LDS. act: 0 = none, 1 = relu.  acc: 1 = accumulate into existing out.
// ---------------------------------------------------------------------------
template <int K>
__global__ void __launch_bounds__(256)
gemm_bias_act(const float* __restrict__ in, int ldin,
              const float* __restrict__ W, int ldw,
              const float* __restrict__ bias,
              float* __restrict__ out, int ldout,
              int M, int N, int act, int acc) {
  constexpr int SWLD = K + 4;            // padded LDS row (bank-conflict avoid)
  __shared__ float sW[192 * SWLD];       // up to 192 weight rows

  const int tid = threadIdx.x;
  constexpr int chunksPerRow = K / 4;    // float4 chunks per weight row
  const int totalChunks = N * chunksPerRow;

#if HAVE_ASYNC_LDS
  for (int j = tid; j < totalChunks; j += blockDim.x) {
    const int row = j / chunksPerRow;
    const int off = (j % chunksPerRow) * 4;
    const float* src = W + (size_t)row * ldw + off;
    float* dst = &sW[row * SWLD + off];
    __builtin_amdgcn_global_load_async_to_lds_b128((gv4i_p)src, (lv4i_p)dst, 0, 0);
  }
  __builtin_amdgcn_s_wait_asynccnt(0);
  __syncthreads();
#else
  for (int j = tid; j < totalChunks; j += blockDim.x) {
    const int row = j / chunksPerRow;
    const int off = (j % chunksPerRow) * 4;
    const float4* src = reinterpret_cast<const float4*>(W + (size_t)row * ldw + off);
    *reinterpret_cast<float4*>(&sW[row * SWLD + off]) = *src;
  }
  __syncthreads();
#endif

  const int w  = (blockIdx.x * blockDim.x + threadIdx.x) >> 5;
  const int m0 = w << 4;
  if (m0 >= M) return;
  const int lane = threadIdx.x & 31;
  const int nlo = lane & 15, nhi = lane >> 4;

  v16bf afr[K / 32];
#pragma unroll
  for (int kk = 0; kk < K / 32; ++kk)
    afr[kk] = ld_frag_f32(in, ldin, m0, kk * 32);

  for (int nt = 0; nt < N; nt += 16) {
    v8f c = {};
    if (acc) {
#pragma unroll
      for (int r = 0; r < 8; ++r)
        c[r] = out[(size_t)(m0 + r + 8 * nhi) * ldout + nt + nlo];
    }
#pragma unroll
    for (int kk = 0; kk < K / 32; ++kk) {
      v16bf b = ld_frag_ldsf(sW, SWLD, nt, kk * 32);
      c = wmma_bf16(afr[kk], b, c);
    }
    const float bv = bias ? bias[nt + nlo] : 0.0f;
#pragma unroll
    for (int r = 0; r < 8; ++r) {
      float v = c[r] + bv;
      if (act == 1) v = fmaxf(v, 0.0f);
      out[(size_t)(m0 + r + 8 * nhi) * ldout + nt + nlo] = v;
    }
  }
}

// ---------------------------------------------------------------------------
// GRU scan: batch-parallel recurrence. gi = precomputed x@Wih^T + bih [L,B,192].
// Each wave owns 16 batch rows for all L steps; h kept in C-layout registers;
// gh = h @ Whh^T via WMMA with Whh staged once in LDS (bf16).
// store_prev=0: out[t] = h_{t+1} (inf GRU d_seq). store_prev=1: out[t] = h_t,
// hn = h_L (gen GRU h_seq / hn).
// ---------------------------------------------------------------------------
#define GRU_WAVES 8
#define SH_LD 68  // padded float row (bank-conflict avoidance)
#define SW_LD 72  // padded bf16 row

__global__ void __launch_bounds__(256)
gru_scan(const float* __restrict__ gi, const float* __restrict__ Whh,
         const float* __restrict__ bhh, float* __restrict__ out,
         float* __restrict__ hn, int Lsteps, int Btot, int store_prev) {
  __shared__ bf16  sW[192 * SW_LD];
  __shared__ float sH[GRU_WAVES][16 * SH_LD];

  const int tid  = threadIdx.x;
  const int lane = tid & 31;
  const int wv   = tid >> 5;
  const int nlo = lane & 15, nhi = lane >> 4;

  // Stage Whh [192,64] -> LDS bf16 (cooperative, once)
  for (int i = tid; i < 192 * 64; i += blockDim.x) {
    int r = i >> 6, c = i & 63;
    sW[r * SW_LD + c] = (bf16)Whh[i];
  }
  __syncthreads();

  const int b0 = (blockIdx.x * GRU_WAVES + wv) * 16;
  float* hrow = sH[wv];

  v8f h[4];
#pragma unroll
  for (int f = 0; f < 4; ++f) h[f] = (v8f){};

  for (int t = 0; t < Lsteps; ++t) {
    const size_t base = (size_t)t * Btot + b0;

    if (store_prev) {
#pragma unroll
      for (int f = 0; f < 4; ++f)
#pragma unroll
        for (int r = 0; r < 8; ++r)
          out[(base + r + 8 * nhi) * 64 + f * 16 + nlo] = h[f][r];
    }

    if (t + 1 < Lsteps)  // global_prefetch of next step's gi rows
      __builtin_prefetch(&gi[((size_t)(t + 1) * Btot + b0 + nhi * 8) * 192 + nlo], 0, 1);

    // Stage h (C-layout regs) -> LDS row-major, then re-load as A-frags.
#pragma unroll
    for (int f = 0; f < 4; ++f)
#pragma unroll
      for (int r = 0; r < 8; ++r)
        hrow[(r + 8 * nhi) * SH_LD + f * 16 + nlo] = h[f][r];
    __syncthreads();

    v16bf a0 = ld_frag_ldsf(hrow, SH_LD, 0, 0);
    v16bf a1 = ld_frag_ldsf(hrow, SH_LD, 0, 32);

    float rg[4][8], zg[4][8];
    // reset gates: cols 0..63
#pragma unroll
    for (int f = 0; f < 4; ++f) {
      v16bf bfr0 = ld_frag_ldsbf(sW, SW_LD, f * 16, 0);
      v16bf bfr1 = ld_frag_ldsbf(sW, SW_LD, f * 16, 32);
      v8f c = {};
      c = wmma_bf16(a0, bfr0, c);
      c = wmma_bf16(a1, bfr1, c);
      const float bb = bhh[f * 16 + nlo];
#pragma unroll
      for (int e = 0; e < 8; ++e) {
        float giv = gi[(base + e + 8 * nhi) * 192 + f * 16 + nlo];
        rg[f][e] = sigmoidf_(giv + c[e] + bb);
      }
    }
    // update gates: cols 64..127
#pragma unroll
    for (int f = 0; f < 4; ++f) {
      v16bf bfr0 = ld_frag_ldsbf(sW, SW_LD, 64 + f * 16, 0);
      v16bf bfr1 = ld_frag_ldsbf(sW, SW_LD, 64 + f * 16, 32);
      v8f c = {};
      c = wmma_bf16(a0, bfr0, c);
      c = wmma_bf16(a1, bfr1, c);
      const float bb = bhh[64 + f * 16 + nlo];
#pragma unroll
      for (int e = 0; e < 8; ++e) {
        float giv = gi[(base + e + 8 * nhi) * 192 + 64 + f * 16 + nlo];
        zg[f][e] = sigmoidf_(giv + c[e] + bb);
      }
    }
    // candidate + state update: cols 128..191
#pragma unroll
    for (int f = 0; f < 4; ++f) {
      v16bf bfr0 = ld_frag_ldsbf(sW, SW_LD, 128 + f * 16, 0);
      v16bf bfr1 = ld_frag_ldsbf(sW, SW_LD, 128 + f * 16, 32);
      v8f c = {};
      c = wmma_bf16(a0, bfr0, c);
      c = wmma_bf16(a1, bfr1, c);
      const float bb = bhh[128 + f * 16 + nlo];
#pragma unroll
      for (int e = 0; e < 8; ++e) {
        float giv = gi[(base + e + 8 * nhi) * 192 + 128 + f * 16 + nlo];
        float nn  = tanhf(giv + rg[f][e] * (c[e] + bb));
        h[f][e] = (1.0f - zg[f][e]) * nn + zg[f][e] * h[f][e];
      }
      if (!store_prev) {
#pragma unroll
        for (int e = 0; e < 8; ++e)
          out[(base + e + 8 * nhi) * 64 + f * 16 + nlo] = h[f][e];
      }
    }
    __syncthreads();
  }

  if (hn) {
#pragma unroll
    for (int f = 0; f < 4; ++f)
#pragma unroll
      for (int r = 0; r < 8; ++r)
        hn[(size_t)(b0 + r + 8 * nhi) * 64 + f * 16 + nlo] = h[f][r];
  }
}

// ---------------------------------------------------------------------------
// Elementwise kernels
// ---------------------------------------------------------------------------
__global__ void gated_combine(const float* __restrict__ g1, const float* __restrict__ g2,
                              float* __restrict__ out, size_t n) {
  size_t i = (size_t)blockIdx.x * blockDim.x + threadIdx.x;
  size_t stride = (size_t)gridDim.x * blockDim.x;
  for (; i < n; i += stride)
    out[i] = fmaxf(g1[i], 0.0f) * sigmoidf_(g2[i]);
}

__global__ void reparam_sample(const float* __restrict__ mu, const float* __restrict__ ls,
                               const float* __restrict__ eps, float* __restrict__ out,
                               size_t n) {
  size_t i = (size_t)blockIdx.x * blockDim.x + threadIdx.x;
  size_t stride = (size_t)gridDim.x * blockDim.x;
  for (; i < n; i += stride)
    out[i] = mu[i] + __expf(0.5f * ls[i]) * eps[i];
}

// ---------------------------------------------------------------------------
// Host orchestration
// ---------------------------------------------------------------------------
extern "C" void kernel_launch(void* const* d_in, const int* in_sizes, int n_in,
                              void* d_out, int out_size, void* d_ws, size_t ws_size,
                              hipStream_t stream) {
  (void)in_sizes; (void)n_in; (void)out_size; (void)ws_size;
  const int L = 512, B = 1024;
  const size_t LB  = (size_t)L * B;
  const size_t S32 = LB * 32;
  const size_t S64 = LB * 64;

  const float* ext = (const float*)d_in[0];
  const float* obs = (const float*)d_in[1];
  const float* eps = (const float*)d_in[2];
  const float* pu_W1 = (const float*)d_in[3],  * pu_b1 = (const float*)d_in[4];
  const float* pu_W2 = (const float*)d_in[5],  * pu_b2 = (const float*)d_in[6];
  const float* px_W1 = (const float*)d_in[7],  * px_b1 = (const float*)d_in[8];
  const float* px_W2 = (const float*)d_in[9],  * px_b2 = (const float*)d_in[10];
  const float* pz_W1 = (const float*)d_in[11], * pz_b1 = (const float*)d_in[12];
  const float* pz_W2 = (const float*)d_in[13], * pz_b2 = (const float*)d_in[14];
  const float* po_W1 = (const float*)d_in[15], * po_b1 = (const float*)d_in[16];
  const float* po_W2 = (const float*)d_in[17], * po_b2 = (const float*)d_in[18];
  const float* po_Wmu = (const float*)d_in[19], * po_bmu = (const float*)d_in[20];
  const float* po_Wls = (const float*)d_in[21], * po_bls = (const float*)d_in[22];
  const float* inf_Wih = (const float*)d_in[23], * inf_Whh = (const float*)d_in[24];
  const float* inf_bih = (const float*)d_in[25], * inf_bhh = (const float*)d_in[26];
  const float* gen_Wih = (const float*)d_in[27], * gen_Whh = (const float*)d_in[28];
  const float* gen_bih = (const float*)d_in[29], * gen_bhh = (const float*)d_in[30];

  float* outp = (float*)d_out;
  float* zmu  = outp;                    // [L,B,32]
  float* zls  = outp + S32;              // [L,B,32]
  float* samp = outp + 2 * S32;          // [L,B,32]
  float* hseq = outp + 3 * S32;          // [L,B,64]
  float* xemb = outp + 3 * S32 + S64;    // [L,B,64]
  float* hnp  = outp + 3 * S32 + 2 * S64;// [B,64]

  float* ws    = (float*)d_ws;
  float* u_emb = ws;                 // S64
  float* tmp   = ws + S64;           // S64 (reused scratch)
  float* dzbuf = ws + 2 * S64;       // S64: d_seq, later z_emb
  float* gibuf = ws + 3 * S64;       // 3*S64 = [L,B,192]; also g1/g2 slots
  float* g1 = gibuf;
  float* g2 = gibuf + S64;

  const int M = (int)LB;
  const dim3 blk(256);
  const dim3 gemm_grid((unsigned)((M / 16) * 32 / 256));
  const dim3 gru_grid((unsigned)(B / (16 * GRU_WAVES)));
  const dim3 ew_grid(4096);

  // --- parallel embeddings -------------------------------------------------
  gemm_bias_act<32><<<gemm_grid, blk, 0, stream>>>(ext, 32, pu_W1, 32, pu_b1, tmp, 64, M, 64, 1, 0);
  gemm_bias_act<64><<<gemm_grid, blk, 0, stream>>>(tmp, 64, pu_W2, 64, pu_b2, u_emb, 64, M, 64, 1, 0);
  gemm_bias_act<32><<<gemm_grid, blk, 0, stream>>>(obs, 32, px_W1, 32, px_b1, tmp, 64, M, 64, 1, 0);
  gemm_bias_act<64><<<gemm_grid, blk, 0, stream>>>(tmp, 64, px_W2, 64, px_b2, xemb, 64, M, 64, 1, 0);

  // --- inference GRU: input projection + batch-parallel scan ---------------
  gemm_bias_act<64><<<gemm_grid, blk, 0, stream>>>(xemb, 64, inf_Wih, 64, inf_bih, gibuf, 192, M, 192, 0, 0);
  gru_scan<<<gru_grid, blk, 0, stream>>>(gibuf, inf_Whh, inf_bhh, dzbuf, nullptr, L, B, 0);

  // --- posterior DBlock + reparameterized sample ---------------------------
  gemm_bias_act<64><<<gemm_grid, blk, 0, stream>>>(dzbuf, 64, po_W1, 64, po_b1, g1, 64, M, 64, 0, 0);
  gemm_bias_act<64><<<gemm_grid, blk, 0, stream>>>(dzbuf, 64, po_W2, 64, po_b2, g2, 64, M, 64, 0, 0);
  gated_combine<<<ew_grid, blk, 0, stream>>>(g1, g2, tmp, S64);
  gemm_bias_act<64><<<gemm_grid, blk, 0, stream>>>(tmp, 64, po_Wmu, 64, po_bmu, zmu, 32, M, 32, 0, 0);
  gemm_bias_act<64><<<gemm_grid, blk, 0, stream>>>(tmp, 64, po_Wls, 64, po_bls, zls, 32, M, 32, 0, 0);
  reparam_sample<<<ew_grid, blk, 0, stream>>>(zmu, zls, eps, samp, S32);

  // --- z embedding ---------------------------------------------------------
  gemm_bias_act<32><<<gemm_grid, blk, 0, stream>>>(samp, 32, pz_W1, 32, pz_b1, tmp, 64, M, 64, 1, 0);
  gemm_bias_act<64><<<gemm_grid, blk, 0, stream>>>(tmp, 64, pz_W2, 64, pz_b2, dzbuf, 64, M, 64, 1, 0);

  // --- generative GRU: concat(z_emb,u_emb) @ Wih^T as two accumulating GEMMs
  gemm_bias_act<64><<<gemm_grid, blk, 0, stream>>>(dzbuf, 64, gen_Wih, 128, gen_bih, gibuf, 192, M, 192, 0, 0);
  gemm_bias_act<64><<<gemm_grid, blk, 0, stream>>>(u_emb, 64, gen_Wih + 64, 128, nullptr, gibuf, 192, M, 192, 0, 1);
  gru_scan<<<gru_grid, blk, 0, stream>>>(gibuf, gen_Whh, gen_bhh, hseq, hnp, L, B, 1);
}
